// NNConvPair_42021960024104
// MI455X (gfx1250) — matrix operand order repaired and checked by reading, not compiled, not run on hardware
//
#include <hip/hip_runtime.h>
#include <math.h>

// Problem constants (from the reference)
#define NNODE 40000
#define NEDGE 80000
#define NB    256
#define C     32      // NNF == EMB == 32 (all conv layers are 32->32)
#define NEFD  16

typedef __attribute__((ext_vector_type(16))) __bf16 v16bf;
typedef __attribute__((ext_vector_type(8)))  float  v8f;

// ---------------- small helpers ----------------
__device__ __forceinline__ unsigned fenc(float f) {
  unsigned b = __float_as_uint(f);
  return (b & 0x80000000u) ? ~b : (b | 0x80000000u);   // monotone float->uint
}
__device__ __forceinline__ float fdec(unsigned u) {
  unsigned b = (u & 0x80000000u) ? (u ^ 0x80000000u) : ~u;
  return __uint_as_float(b);
}

__global__ void zero_f32(float* p, int n) {
  int i = blockIdx.x * blockDim.x + threadIdx.x;
  if (i < n) p[i] = 0.0f;
}

// ---------------- attention pooling on raw node features ----------------
__global__ void att_g_kernel(const float* __restrict__ x,
                             const float* __restrict__ W1, const float* __restrict__ b1,
                             const float* __restrict__ W2, const float* __restrict__ b2,
                             float* __restrict__ g, int n) {
  int i = blockIdx.x * blockDim.x + threadIdx.x;
  if (i >= n) return;
  const float* xi = x + (size_t)i * C;
  float acc = b2[0];
  for (int j = 0; j < C; ++j) {
    float h = b1[j];
    for (int k = 0; k < C; ++k) h += xi[k] * W1[k * C + j];
    acc += fmaxf(h, 0.0f) * W2[j];      // att_W2 is [C,1]
  }
  g[i] = acc;
}

__global__ void seg_max_enc(const float* __restrict__ g, const int* __restrict__ batch,
                            unsigned* __restrict__ m, int n) {
  int i = blockIdx.x * blockDim.x + threadIdx.x;
  if (i >= n) return;
  atomicMax(&m[batch[i]], fenc(g[i]));
}

__global__ void att_exp_kernel(float* __restrict__ g, const int* __restrict__ batch,
                               const unsigned* __restrict__ m, float* __restrict__ s, int n) {
  int i = blockIdx.x * blockDim.x + threadIdx.x;
  if (i >= n) return;
  int b = batch[i];
  float e = expf(g[i] - fdec(m[b]));
  g[i] = e;
  atomicAdd(&s[b], e);
}

__global__ void att_pool_kernel(const float* __restrict__ g, const float* __restrict__ s,
                                const int* __restrict__ batch, const float* __restrict__ x,
                                float* __restrict__ pool, int n) {
  int t = blockIdx.x * blockDim.x + threadIdx.x;
  if (t >= n * C) return;
  int i = t >> 5, c = t & 31;
  int b = batch[i];
  float gate = g[i] / s[b];
  atomicAdd(&pool[b * C + c], gate * x[(size_t)i * C + c]);
}

// ---------------- edge MLP first layer: h = relu(edge_attr @ W1 + b1) ----------------
__global__ void edge_h_kernel(const float* __restrict__ ea,
                              const float* __restrict__ W1, const float* __restrict__ b1,
                              float* __restrict__ h, int ne) {
  int t = blockIdx.x * blockDim.x + threadIdx.x;
  if (t >= ne * 16) return;
  int e = t >> 4, k = t & 15;
  const float* a = ea + (size_t)e * NEFD;
  float acc = b1[k];
  for (int i = 0; i < NEFD; ++i) acc += a[i] * W1[i * 16 + k];
  h[t] = fmaxf(acc, 0.0f);
}

// ---------------- fused NNConv message+scatter (the WMMA centerpiece) ----------------
// msg_e = sum_k h_e[k]*(x_src @ W2k) + x_src @ Tb, with W2k/Tb shared 32x32 matrices
// (theta is never materialized: saves ~3.2 GB of HBM traffic vs the naive form).
// One wave = one 16-edge tile. A operand = h-scaled gathered x rows (bf16, scaled with
// packed bf16 muls), B operands = W2k column halves preswizzled in LDS, f32 accumulation,
// then one global f32 atomicAdd per output element into agg[dst].
// The A operand for km+1 is produced BEFORE km's WMMAs are issued: the packed-bf16
// scaling stream co-executes with the matrix pipe and the WMMA->VALU WAR hazard on
// a_cur is pushed one iteration away (removes the v_nop stalls seen previously).
__global__ __launch_bounds__(256)
void conv_edge_wmma(const float* __restrict__ x, const int* __restrict__ src,
                    const int* __restrict__ dst, const float* __restrict__ h,
                    const float* __restrict__ W2, const float* __restrict__ b2,
                    float* __restrict__ agg, int ntiles) {
  // 17 matrices (16 x W2k + bias matrix), each 32x32 bf16, preswizzled to the
  // per-lane B-operand layout: lane = (o&15) + 16*(k>>4), elem = k&15, half = o>>4.
  __shared__ __align__(32) __bf16 lw[17 * 1024];
  for (int t = threadIdx.x; t < 17 * 1024; t += blockDim.x) {
    int mat = t >> 10;
    int r = t & 1023;        // r = k*32 + o
    int k = r >> 5, o = r & 31;
    float v = (mat < 16) ? W2[mat * 1024 + r] : b2[r];
    lw[mat * 1024 + (o >> 4) * 512 + (((o & 15) + ((k >> 4) << 4)) << 4) + (k & 15)] = (__bf16)v;
  }
  __syncthreads();

  int wid = threadIdx.x >> 5, lane = threadIdx.x & 31;
  int tile = blockIdx.x * 8 + wid;
  if (tile >= ntiles) return;
  int half = lane >> 4, r = lane & 15;
  int e = tile * 16 + r;                 // E is an exact multiple of 16
  int s = src[e];
  const float* xr = x + (size_t)s * C;

  // A operand gather (contiguous K order: lanes<16 K 0..15; lanes>=16 K 16..31)
  v16bf ab;
#pragma unroll
  for (int i = 0; i < 8; ++i) ab[i] = (__bf16)xr[half * 16 + i];
#pragma unroll
  for (int i = 0; i < 8; ++i) ab[8 + i] = (__bf16)xr[half * 16 + 8 + i];

  // per-edge hidden activations as bf16 scales
  const float* hr = h + (size_t)e * 16;
  __bf16 hb[16];
#pragma unroll
  for (int i = 0; i < 16; ++i) hb[i] = (__bf16)hr[i];

  v8f acc0 = {}, acc1 = {};

  // scale A rows by h[km] with packed bf16 muls; km==16 is the bias matrix (scale 1)
  auto scaleA = [&](int km) -> v16bf {
    if (km >= 16) return ab;
    v16bf sv;
#pragma unroll
    for (int i = 0; i < 16; ++i) sv[i] = hb[km];
    return ab * sv;
  };

  v16bf a_cur = scaleA(0);
#pragma unroll
  for (int km = 0; km < 17; ++km) {
    v16bf a_next = (km < 16) ? scaleA(km + 1) : a_cur;   // software pipeline
    v16bf b0 = *(const v16bf*)(lw + km * 1024 + lane * 16);
    v16bf b1 = *(const v16bf*)(lw + km * 1024 + 512 + lane * 16);
    acc0 = __builtin_amdgcn_wmma_f32_16x16x32_bf16(false, a_cur, false, b0, (short)0, acc0, false, false);
    acc1 = __builtin_amdgcn_wmma_f32_16x16x32_bf16(false, a_cur, false, b1, (short)0, acc1, false, false);
    a_cur = a_next;
  }

  // Scatter: C/D layout => lane<16: rows 0..7, lane>=16: rows 8..15; col = lane&15 (+16 for acc1)
  int n0 = lane & 15;
#pragma unroll
  for (int j = 0; j < 8; ++j) {
    int m = tile * 16 + ((lane < 16) ? j : (8 + j));
    int d = dst[m];
    atomicAdd(&agg[(size_t)d * C + n0], acc0[j]);
    atomicAdd(&agg[(size_t)d * C + 16 + n0], acc1[j]);
  }
}

// ---------------- node update: x_out = relu(agg + x_in @ root + bias), WMMA with C=agg ----------------
__global__ __launch_bounds__(256)
void node_update_wmma(const float* __restrict__ xin, const float* __restrict__ agg,
                      const float* __restrict__ root, const float* __restrict__ bias,
                      float* __restrict__ xout, int ntiles) {
  __shared__ __align__(32) __bf16 lw[1024];
  for (int t = threadIdx.x; t < 1024; t += blockDim.x) {
    int k = t >> 5, o = t & 31;
    lw[(o >> 4) * 512 + (((o & 15) + ((k >> 4) << 4)) << 4) + (k & 15)] = (__bf16)root[t];
  }
  __syncthreads();

  int wid = threadIdx.x >> 5, lane = threadIdx.x & 31;
  int tile = blockIdx.x * 8 + wid;
  if (tile >= ntiles) return;
  int half = lane >> 4, r = lane & 15;
  int node = tile * 16 + r;              // N is an exact multiple of 16
  const float* xr = xin + (size_t)node * C;

  v16bf a;
#pragma unroll
  for (int i = 0; i < 8; ++i) a[i] = (__bf16)xr[half * 16 + i];
#pragma unroll
  for (int i = 0; i < 8; ++i) a[8 + i] = (__bf16)xr[half * 16 + 8 + i];

  int n0 = lane & 15;
  v8f c0, c1;
#pragma unroll
  for (int j = 0; j < 8; ++j) {
    int m = tile * 16 + ((lane < 16) ? j : (8 + j));
    c0[j] = agg[(size_t)m * C + n0];
    c1[j] = agg[(size_t)m * C + 16 + n0];
  }
  v16bf b0 = *(const v16bf*)(lw + lane * 16);
  v16bf b1 = *(const v16bf*)(lw + 512 + lane * 16);
  c0 = __builtin_amdgcn_wmma_f32_16x16x32_bf16(false, a, false, b0, (short)0, c0, false, false);
  c1 = __builtin_amdgcn_wmma_f32_16x16x32_bf16(false, a, false, b1, (short)0, c1, false, false);

  float bn0 = bias[n0], bn1 = bias[16 + n0];
#pragma unroll
  for (int j = 0; j < 8; ++j) {
    int m = tile * 16 + ((lane < 16) ? j : (8 + j));
    xout[(size_t)m * C + n0]      = fmaxf(c0[j] + bn0, 0.0f);
    xout[(size_t)m * C + 16 + n0] = fmaxf(c1[j] + bn1, 0.0f);
  }
}

// ---------------- global max pool (values are >=0 after relu => uint atomicMax works) ----------------
__global__ void seg_maxpool(const float* __restrict__ x, const int* __restrict__ batch,
                            unsigned* __restrict__ gm, int n) {
  int t = blockIdx.x * blockDim.x + threadIdx.x;
  if (t >= n * C) return;
  int i = t >> 5, c = t & 31;
  atomicMax(&gm[batch[i] * C + c], __float_as_uint(x[t]));
}

// ---------------- head: concat -> Linear(128,128) -> Linear(128,1) ----------------
__global__ void head1_kernel(const unsigned* __restrict__ gmp, const unsigned* __restrict__ gmd,
                             const float* __restrict__ ap, const float* __restrict__ ad,
                             const float* __restrict__ W0, const float* __restrict__ b0,
                             float* __restrict__ t) {
  int b = blockIdx.x, j = threadIdx.x;   // 128 threads
  __shared__ float cat[128];
  if (j < 32)       cat[j] = __uint_as_float(gmp[b * 32 + j]);
  else if (j < 64)  cat[j] = __uint_as_float(gmd[b * 32 + (j - 32)]);
  else if (j < 96)  cat[j] = ap[b * 32 + (j - 64)];
  else              cat[j] = ad[b * 32 + (j - 96)];
  __syncthreads();
  float acc = b0[j];
  for (int i = 0; i < 128; ++i) acc += cat[i] * W0[i * 128 + j];
  t[b * 128 + j] = acc;
}

__global__ void head2_kernel(const float* __restrict__ t, const float* __restrict__ W1,
                             const float* __restrict__ b1, float* __restrict__ out) {
  int b = blockIdx.x, j = threadIdx.x;   // 128 threads
  __shared__ float red[128];
  red[j] = t[b * 128 + j] * W1[j];
  __syncthreads();
  for (int off = 64; off > 0; off >>= 1) {
    if (j < off) red[j] += red[j + off];
    __syncthreads();
  }
  if (j == 0) out[b] = red[0] + b1[0];
}

// ---------------- host orchestration ----------------
// JAX pytree flattening order (dict keys sorted): inputs 0..7, then params:
//   side 'd' (18 leaves), lin_W0, lin_W1, lin_b0, lin_b1, side 'p' (18 leaves)
// Side leaf order: att_W1, att_W2, att_b1, att_b2, bias0, biases[0], biases[1],
//   nn0_W1, nn0_W2, nn0_b1, nn0_b2, nn1_W1, nn1_W2, nn1_b1, nn1_b2, root0, roots[0], roots[1]
struct SideP {
  const float *att_W1, *att_W2, *att_b1, *att_b2;
  const float *bias0, *bias1, *bias2;
  const float *nn0_W1, *nn0_W2, *nn0_b1, *nn0_b2;
  const float *nn1_W1, *nn1_W2, *nn1_b1, *nn1_b2;
  const float *root0, *root1, *root2;
};
static SideP load_side(void* const* d_in, int base) {
  SideP s;
  s.att_W1 = (const float*)d_in[base + 0];  s.att_W2 = (const float*)d_in[base + 1];
  s.att_b1 = (const float*)d_in[base + 2];  s.att_b2 = (const float*)d_in[base + 3];
  s.bias0  = (const float*)d_in[base + 4];  s.bias1  = (const float*)d_in[base + 5];
  s.bias2  = (const float*)d_in[base + 6];
  s.nn0_W1 = (const float*)d_in[base + 7];  s.nn0_W2 = (const float*)d_in[base + 8];
  s.nn0_b1 = (const float*)d_in[base + 9];  s.nn0_b2 = (const float*)d_in[base + 10];
  s.nn1_W1 = (const float*)d_in[base + 11]; s.nn1_W2 = (const float*)d_in[base + 12];
  s.nn1_b1 = (const float*)d_in[base + 13]; s.nn1_b2 = (const float*)d_in[base + 14];
  s.root0  = (const float*)d_in[base + 15]; s.root1  = (const float*)d_in[base + 16];
  s.root2  = (const float*)d_in[base + 17];
  return s;
}

static void run_side(const float* x0, const float* eattr, const int* eidx, const int* batch,
                     const SideP& P, float* h0, float* h1, float* xA, float* xB, float* agg,
                     float* g, unsigned* m, float* s, float* attp, unsigned* gmax,
                     hipStream_t stream) {
  const int* src = eidx;
  const int* dst = eidx + NEDGE;
  const int NT_E = NEDGE / 16;   // 5000 tiles
  const int NT_N = NNODE / 16;   // 2500 tiles

  // attention pooling on RAW features
  zero_f32<<<1, NB, 0, stream>>>((float*)m, NB);
  zero_f32<<<1, NB, 0, stream>>>(s, NB);
  zero_f32<<<(NB * C + 255) / 256, 256, 0, stream>>>(attp, NB * C);
  zero_f32<<<(NB * C + 255) / 256, 256, 0, stream>>>((float*)gmax, NB * C);
  att_g_kernel<<<(NNODE + 255) / 256, 256, 0, stream>>>(x0, P.att_W1, P.att_b1, P.att_W2, P.att_b2, g, NNODE);
  seg_max_enc<<<(NNODE + 255) / 256, 256, 0, stream>>>(g, batch, m, NNODE);
  att_exp_kernel<<<(NNODE + 255) / 256, 256, 0, stream>>>(g, batch, m, s, NNODE);
  att_pool_kernel<<<(NNODE * C + 255) / 256, 256, 0, stream>>>(g, s, batch, x0, attp, NNODE);

  // edge MLP hidden activations (nn0 and shared nn1)
  edge_h_kernel<<<(NEDGE * 16 + 255) / 256, 256, 0, stream>>>(eattr, P.nn0_W1, P.nn0_b1, h0, NEDGE);
  edge_h_kernel<<<(NEDGE * 16 + 255) / 256, 256, 0, stream>>>(eattr, P.nn1_W1, P.nn1_b1, h1, NEDGE);

  // layer 0
  zero_f32<<<(NNODE * C + 255) / 256, 256, 0, stream>>>(agg, NNODE * C);
  conv_edge_wmma<<<NT_E / 8, 256, 0, stream>>>(x0, src, dst, h0, P.nn0_W2, P.nn0_b2, agg, NT_E);
  node_update_wmma<<<(NT_N + 7) / 8, 256, 0, stream>>>(x0, agg, P.root0, P.bias0, xA, NT_N);
  // layer 1
  zero_f32<<<(NNODE * C + 255) / 256, 256, 0, stream>>>(agg, NNODE * C);
  conv_edge_wmma<<<NT_E / 8, 256, 0, stream>>>(xA, src, dst, h1, P.nn1_W2, P.nn1_b2, agg, NT_E);
  node_update_wmma<<<(NT_N + 7) / 8, 256, 0, stream>>>(xA, agg, P.root1, P.bias1, xB, NT_N);
  // layer 2
  zero_f32<<<(NNODE * C + 255) / 256, 256, 0, stream>>>(agg, NNODE * C);
  conv_edge_wmma<<<NT_E / 8, 256, 0, stream>>>(xB, src, dst, h1, P.nn1_W2, P.nn1_b2, agg, NT_E);
  node_update_wmma<<<(NT_N + 7) / 8, 256, 0, stream>>>(xB, agg, P.root2, P.bias2, xA, NT_N);

  seg_maxpool<<<(NNODE * C + 255) / 256, 256, 0, stream>>>(xA, batch, gmax, NNODE);
}

extern "C" void kernel_launch(void* const* d_in, const int* in_sizes, int n_in,
                              void* d_out, int out_size, void* d_ws, size_t ws_size,
                              hipStream_t stream) {
  (void)in_sizes; (void)n_in; (void)out_size; (void)ws_size;
  const float* x_p   = (const float*)d_in[0];
  const float* x_d   = (const float*)d_in[1];
  const float* ea_p  = (const float*)d_in[2];
  const float* ea_d  = (const float*)d_in[3];
  const int*   ei_p  = (const int*)d_in[4];
  const int*   ei_d  = (const int*)d_in[5];
  const int*   bat_p = (const int*)d_in[6];
  const int*   bat_d = (const int*)d_in[7];
  SideP Pd = load_side(d_in, 8);
  const float* lin_W0 = (const float*)d_in[26];
  const float* lin_W1 = (const float*)d_in[27];
  const float* lin_b0 = (const float*)d_in[28];
  const float* lin_b1 = (const float*)d_in[29];
  SideP Pp = load_side(d_in, 30);

  // workspace layout
  char* w = (char*)d_ws;
  const size_t HB = (size_t)NEDGE * 16 * 4;   // 5.12 MB
  const size_t XB = (size_t)NNODE * C * 4;    // 5.12 MB
  float* h0_p = (float*)(w);            float* h1_p = (float*)(w + HB);
  float* h0_d = (float*)(w + 2 * HB);   float* h1_d = (float*)(w + 3 * HB);
  char* q = w + 4 * HB;
  float* xA_p = (float*)(q);            float* xB_p = (float*)(q + XB);
  float* ag_p = (float*)(q + 2 * XB);
  float* xA_d = (float*)(q + 3 * XB);   float* xB_d = (float*)(q + 4 * XB);
  float* ag_d = (float*)(q + 5 * XB);
  char* u = q + 6 * XB;
  float* g_p = (float*)(u);                       float* g_d = (float*)(u + (size_t)NNODE * 4);
  char* v = u + 2 * (size_t)NNODE * 4;
  unsigned* m_p = (unsigned*)(v);                 float* s_p = (float*)(v + 1024);
  unsigned* m_d = (unsigned*)(v + 2048);          float* s_d = (float*)(v + 3072);
  float* attp_p = (float*)(v + 4096);             float* attp_d = (float*)(v + 4096 + NB * C * 4);
  unsigned* gm_p = (unsigned*)(v + 4096 + 2 * NB * C * 4);
  unsigned* gm_d = (unsigned*)(v + 4096 + 3 * NB * C * 4);
  float* headt   = (float*)(v + 4096 + 4 * NB * C * 4);

  run_side(x_p, ea_p, ei_p, bat_p, Pp, h0_p, h1_p, xA_p, xB_p, ag_p, g_p, m_p, s_p, attp_p, gm_p, stream);
  run_side(x_d, ea_d, ei_d, bat_d, Pd, h0_d, h1_d, xA_d, xB_d, ag_d, g_d, m_d, s_d, attp_d, gm_d, stream);

  head1_kernel<<<NB, 128, 0, stream>>>(gm_p, gm_d, attp_p, attp_d, lin_W0, lin_b0, headt);
  head2_kernel<<<NB, 128, 0, stream>>>(headt, lin_W1, lin_b1, (float*)d_out);
}
